// KernelLayer_65481071400087
// MI455X (gfx1250) — matrix operand order repaired and checked by reading, not compiled, not run on hardware
//
#include <hip/hip_runtime.h>

// RBF Gram matrix: out[i][j] = exp(-gamma * max(||x_i||^2 + ||s_j||^2 - 2 x_i.s_j, 0))
// N = M = 8192, D = 256. bf16 hi/lo split GEMM on v_wmma_f32_16x16x32_bf16,
// double-buffered LDS staging via global_load_async_to_lds_b128 (ASYNCcnt).

#define NN 8192
#define MM 8192
#define DD 256
#define GAMMA (1.0f / 256.0f)

typedef __attribute__((ext_vector_type(16))) __bf16 v16bf;
typedef __attribute__((ext_vector_type(8)))  __bf16 v8bf;
typedef __attribute__((ext_vector_type(8)))  float  v8f;

#define LDS_STRIDE 40   // 32 k-elements + 8 pad (80B rows -> conflict-free b128)
#define TILE_ELEMS (128 * LDS_STRIDE)

// ---------------------------------------------------------------------------
// Prep: per-row squared norm (f32) + bf16 hi/lo decomposition of the inputs.
// One 256-thread block per row; rows [0,N) are x, rows [N,N+M) are support.
// ---------------------------------------------------------------------------
__global__ __launch_bounds__(256) void rbf_prep_kernel(
    const float* __restrict__ x, const float* __restrict__ s,
    __bf16* __restrict__ xhi, __bf16* __restrict__ xlo,
    __bf16* __restrict__ shi, __bf16* __restrict__ slo,
    float* __restrict__ xnorm, float* __restrict__ snorm) {
  int row = blockIdx.x;
  int t = threadIdx.x;

  const float* src;
  __bf16 *dhi, *dlo;
  float* nrm;
  if (row < NN) {
    src = x + (size_t)row * DD;
    dhi = xhi + (size_t)row * DD;
    dlo = xlo + (size_t)row * DD;
    nrm = xnorm + row;
  } else {
    int r = row - NN;
    src = s + (size_t)r * DD;
    dhi = shi + (size_t)r * DD;
    dlo = slo + (size_t)r * DD;
    nrm = snorm + r;
  }

  float v = src[t];
  __bf16 h = (__bf16)v;
  __bf16 l = (__bf16)(v - (float)h);
  dhi[t] = h;
  dlo[t] = l;

  __shared__ float red[256];
  red[t] = v * v;
  __syncthreads();
#pragma unroll
  for (int off = 128; off > 0; off >>= 1) {
    if (t < off) red[t] += red[t + off];
    __syncthreads();
  }
  if (t == 0) *nrm = red[0];
}

// ---------------------------------------------------------------------------
// CDNA5 async memory->LDS staging (GV mode): per-lane 16B transfer, tracked
// with ASYNCcnt, no data VGPRs. LDS dest address = generic addr truncated to
// 32 bits (shared aperture keeps LDS offset in addr[31:0]).
// ---------------------------------------------------------------------------
__device__ __forceinline__ void async_ld_b128(void* lds, const void* gptr) {
  unsigned l = (unsigned)(unsigned long long)lds;
  asm volatile("global_load_async_to_lds_b128 %0, %1, off"
               :: "v"(l), "v"(gptr) : "memory");
}

__device__ __forceinline__ void wait_async0() {
  asm volatile("s_wait_asynccnt 0x0" ::: "memory");
}

// ---------------------------------------------------------------------------
// Fragment loads from LDS per CDNA5 ISA 7.12.2 (wave32, 16-bit operands).
// A (16x32, MxK): lane L holds row M=L&15; lanes<16: K in {0-7,16-23},
//                 lanes>=16: K in {8-15,24-31}.
// B (32x16, KxN): lane L holds col N=L&15; lanes<16: K in {0-7,16-23}, lanes>=16: K in {8-15,24-31}.
// ---------------------------------------------------------------------------
__device__ __forceinline__ v16bf load_frag_a(const __bf16* tile, int r_base, int lane) {
  int row = r_base + (lane & 15);
  int k1 = (lane < 16) ? 0 : 8;
  v8bf p0 = *(const v8bf*)(tile + row * LDS_STRIDE + k1);
  v8bf p1 = *(const v8bf*)(tile + row * LDS_STRIDE + k1 + 16);
  return __builtin_shufflevector(p0, p1, 0, 1, 2, 3, 4, 5, 6, 7, 8, 9, 10, 11, 12, 13, 14, 15);
}

__device__ __forceinline__ v16bf load_frag_b(const __bf16* tile, int n_base, int lane) {
  int row = n_base + (lane & 15);  // tile stores support rows; B[k][n] = s[n][k]
  int k1 = (lane < 16) ? 0 : 8;
  v8bf p0 = *(const v8bf*)(tile + row * LDS_STRIDE + k1);
  v8bf p1 = *(const v8bf*)(tile + row * LDS_STRIDE + k1 + 16);
  return __builtin_shufflevector(p0, p1, 0, 1, 2, 3, 4, 5, 6, 7, 8, 9, 10, 11, 12, 13, 14, 15);
}

__device__ __forceinline__ v8f wmma_bf16(v16bf a, v16bf b, v8f c) {
  return __builtin_amdgcn_wmma_f32_16x16x32_bf16(false, a, false, b, (short)0, c, false, false);
}

// ---------------------------------------------------------------------------
// GEMM + fused epilogue. Block = 128x128 output tile, 256 threads = 8 waves.
// Wave (wm,wn) in 2x4 arrangement owns a 64x32 region = 4x2 WMMA tiles.
// Double-buffered LDS: async loads for tile k+1 overlap WMMAs on tile k.
// ---------------------------------------------------------------------------
__global__ __launch_bounds__(256) void rbf_gemm_kernel(
    const __bf16* __restrict__ xhi, const __bf16* __restrict__ xlo,
    const __bf16* __restrict__ shi, const __bf16* __restrict__ slo,
    const float* __restrict__ xnorm, const float* __restrict__ snorm,
    float* __restrict__ out) {
  // [buf][matrix: 0=Ahi 1=Alo 2=Bhi 3=Blo][128*LDS_STRIDE]
  __shared__ __bf16 smem[2][4][TILE_ELEMS];

  int bn = blockIdx.x;   // column block (support)
  int bm = blockIdx.y;   // row block (x)
  int tid = threadIdx.x;
  int lane = tid & 31;
  int wave = tid >> 5;
  int wm = wave >> 2;    // 0..1 -> 64-row slab
  int wn = wave & 3;     // 0..3 -> 32-col slab

  v8f acc[4][2] = {};

  // Staging geometry: 4 tiles of 128 rows x 32 k bf16 (8 KB each) per buffer;
  // 512 16B chunks/tile, 2 chunks per thread per tile -> 8 async b128 / thread.
  int c0 = tid;               // chunk ids c0, c0+256
  // issue one buffer's loads
  auto issue = [&](int buf, int kk) {
#pragma unroll
    for (int i = 0; i < 2; ++i) {
      int c = c0 + 256 * i;           // 0..511
      int r = c >> 2;                 // row 0..127
      int seg = (c & 3) * 8;          // k sub-offset in bf16 elements
      size_t gA = (size_t)(bm * 128 + r) * DD + kk + seg;
      size_t gB = (size_t)(bn * 128 + r) * DD + kk + seg;
      int l = r * LDS_STRIDE + seg;
      async_ld_b128(&smem[buf][0][l], xhi + gA);
      async_ld_b128(&smem[buf][1][l], xlo + gA);
      async_ld_b128(&smem[buf][2][l], shi + gB);
      async_ld_b128(&smem[buf][3][l], slo + gB);
    }
  };

  issue(0, 0);

  for (int k = 0; k < DD / 32; ++k) {
    int cur = k & 1;
    wait_async0();      // my async loads for buf[cur] have landed in LDS
    __syncthreads();    // everyone's loads landed; everyone done reading buf[1-cur]
    if (k + 1 < DD / 32) issue(1 - cur, (k + 1) * 32);  // overlap with compute

    const __bf16* tAhi = smem[cur][0];
    const __bf16* tAlo = smem[cur][1];
    const __bf16* tBhi = smem[cur][2];
    const __bf16* tBlo = smem[cur][3];

    v16bf ah[4], al[4], bh[2], bl[2];
#pragma unroll
    for (int tm = 0; tm < 4; ++tm) {
      int rb = wm * 64 + tm * 16;
      ah[tm] = load_frag_a(tAhi, rb, lane);
      al[tm] = load_frag_a(tAlo, rb, lane);
    }
#pragma unroll
    for (int tn = 0; tn < 2; ++tn) {
      int nb = wn * 32 + tn * 16;
      bh[tn] = load_frag_b(tBhi, nb, lane);
      bl[tn] = load_frag_b(tBlo, nb, lane);
    }
#pragma unroll
    for (int tm = 0; tm < 4; ++tm) {
#pragma unroll
      for (int tn = 0; tn < 2; ++tn) {
        // cross ~= hi*hi + lo*hi + hi*lo  (ll term negligible)
        acc[tm][tn] = wmma_bf16(ah[tm], bh[tn], acc[tm][tn]);
        acc[tm][tn] = wmma_bf16(al[tm], bh[tn], acc[tm][tn]);
        acc[tm][tn] = wmma_bf16(ah[tm], bl[tn], acc[tm][tn]);
      }
    }
    __syncthreads();    // done reading buf[cur] before next iter refills it
  }

  // Epilogue: C/D layout -> lanes 0-15: M = vgpr, lanes 16-31: M = vgpr + 8;
  // N = lane & 15.
  int row_half = (lane >> 4) << 3;  // 0 or 8
  int col_in_tile = lane & 15;
#pragma unroll
  for (int tm = 0; tm < 4; ++tm) {
    int grow_base = bm * 128 + wm * 64 + tm * 16 + row_half;
#pragma unroll
    for (int tn = 0; tn < 2; ++tn) {
      int gcol = bn * 128 + wn * 32 + tn * 16 + col_in_tile;
      float sn = snorm[gcol];
#pragma unroll
      for (int r = 0; r < 8; ++r) {
        int grow = grow_base + r;
        float d = xnorm[grow] + sn - 2.0f * acc[tm][tn][r];
        d = fmaxf(d, 0.0f);
        out[(size_t)grow * MM + gcol] = __expf(-GAMMA * d);
      }
    }
  }
}

// ---------------------------------------------------------------------------
// Launch. d_ws layout (all 16B aligned):
//   [0,4MB)      xhi   (8192*256 bf16)
//   [4MB,8MB)    xlo
//   [8MB,12MB)   shi
//   [12MB,16MB)  slo
//   [16MB,+32KB) xnorm (8192 f32)
//   [+32KB,+64KB)snorm (8192 f32)
// ---------------------------------------------------------------------------
extern "C" void kernel_launch(void* const* d_in, const int* in_sizes, int n_in,
                              void* d_out, int out_size, void* d_ws, size_t ws_size,
                              hipStream_t stream) {
  const float* x = (const float*)d_in[0];
  const float* s = (const float*)d_in[1];
  float* out = (float*)d_out;

  char* ws = (char*)d_ws;
  const size_t elems = (size_t)NN * DD;  // == MM*DD
  __bf16* xhi = (__bf16*)(ws);
  __bf16* xlo = (__bf16*)(ws + elems * 2);
  __bf16* shi = (__bf16*)(ws + elems * 4);
  __bf16* slo = (__bf16*)(ws + elems * 6);
  float* xnorm = (float*)(ws + elems * 8);
  float* snorm = (float*)(ws + elems * 8 + NN * sizeof(float));

  rbf_prep_kernel<<<NN + MM, 256, 0, stream>>>(x, s, xhi, xlo, shi, slo, xnorm, snorm);

  dim3 grid(MM / 128, NN / 128);
  rbf_gemm_kernel<<<grid, 256, 0, stream>>>(xhi, xlo, shi, slo, xnorm, snorm, out);
}
